// SIT_69406671504084
// MI455X (gfx1250) — matrix operand order repaired and checked by reading, not compiled, not run on hardware
//
#include <hip/hip_runtime.h>
#include <math.h>

// ---------------- model constants ----------------
#define BB 4
#define HH 32
#define DD 64
#define DMODEL 512
#define NHEADS 8
#define HEADDIM 64
#define FFDIM 2048
#define HIDC 128
#define NTOK 8192          // B*H*D tokens
#define QKVDIM 1536

typedef _Float16 half_t;
typedef __attribute__((ext_vector_type(16))) _Float16 v16h;
typedef __attribute__((ext_vector_type(8)))  _Float16 v8h;
typedef __attribute__((ext_vector_type(8)))  float    v8f;
typedef __attribute__((ext_vector_type(4)))  unsigned int v4u;
typedef __attribute__((ext_vector_type(4)))  int      v4i;
typedef __attribute__((ext_vector_type(8)))  int      v8i;

#if defined(__has_builtin)
#  if __has_builtin(__builtin_amdgcn_tensor_load_to_lds)
#    define HAVE_TDM 1
#  endif
#endif
#ifndef HAVE_TDM
#  define HAVE_TDM 0
#endif

// ---- TDM descriptor packing (CDNA5 ISA ch.8, D# group0/group1) ----
// group0: [1:0]=count=1, [63:32]=lds_addr(bytes), [120:64]=global_addr, [127:126]=type=2
__device__ inline v4u tdm_g0(unsigned lds_off, unsigned long long gaddr) {
    v4u g;
    g[0] = 1u;
    g[1] = lds_off;
    g[2] = (unsigned)(gaddr & 0xFFFFFFFFu);
    g[3] = (unsigned)((gaddr >> 32) & 0x01FFFFFFu) | (2u << 30);
    return g;
}
// group1: data_size=2B; 2-D tile: tile_dim0 x tile_dim1, row stride = stride0 elems
__device__ inline v8i tdm_g1(unsigned tdim0, unsigned tdim1,
                             unsigned tile0, unsigned tile1,
                             unsigned long long stride0) {
    v8i g = {};
    g[0] = 0x00010000;                                             // data_size=1 -> 2 bytes
    g[1] = (int)((tdim0 & 0xFFFFu) << 16);                         // tensor_dim0 lo16 @63:48
    g[2] = (int)(((tdim0 >> 16) & 0xFFFFu) | ((tdim1 & 0xFFFFu) << 16));
    g[3] = (int)(((tdim1 >> 16) & 0xFFFFu) | ((tile0 & 0xFFFFu) << 16)); // tile_dim0 @127:112
    g[4] = (int)(tile1 & 0xFFFFu);                                 // tile_dim1; tile_dim2=0
    g[5] = (int)(stride0 & 0xFFFFFFFFull);                         // tensor_dim0_stride lo32
    g[6] = (int)((stride0 >> 32) & 0xFFFFull);                     // hi16
    g[7] = 0;
    return g;
}

// ==================== f32 -> f16 conversion (2 elems/thread) ==========
__global__ void cvt_f16_kernel(const float* __restrict__ in, half_t* __restrict__ out, int n)
{
    int i = (blockIdx.x * blockDim.x + threadIdx.x) * 2;
    if (i >= n) return;
    out[i]     = (half_t)in[i];
    out[i + 1] = (half_t)in[i + 1];
}

// =====================================================================
// Tiled WMMA GEMM:  C[M,N] = A16[M,K] @ W16[N,K]^T + biasScale*bias[N]
// Block = 256 thr (8 waves), block tile 128x128, K-step 32.
// K-step tiles (128x32 f16 each for A and B) staged into LDS by the
// Tensor Data Mover; fragments read via ds_load_b128 per ISA 7.12.2:
//   A 16x32 f16 : lane-> M=L&15, kb=(L>>4)*8 ; halves {kb..kb+7},{kb+16..}
//   B 32x16 f16 : lane-> N=L&15, kb=(L>>4)*8 ; halves {kb..kb+7},{kb+16..}
//   C 16x16 f32 : lane-> N=L&15 ; VGPR r -> M = r + (L>>4)*8
// Each wave: 32x64 sub-tile = 2 A frags x 4 B frags -> 8 WMMA / step.
// =====================================================================
__global__ void gemm_wmma_kernel(const half_t* __restrict__ A,
                                 const half_t* __restrict__ W,
                                 const float* __restrict__ bias,
                                 float* __restrict__ C,
                                 int M, int N, int K, int doRelu, float biasScale)
{
    extern __shared__ __align__(16) _Float16 smem[];
    _Float16* sA = smem;            // 128 x 32 f16  (8 KB)  lds offset 0
    _Float16* sB = smem + 4096;     // 128 x 32 f16  (8 KB)  lds offset 8192

    const int lane = threadIdx.x & 31;
    const int wid  = threadIdx.x >> 5;
    const int bn   = blockIdx.x % (N >> 7);
    const int bm   = blockIdx.x / (N >> 7);
    const int wm   = wid & 3;       // 4 M-strips of 32 rows
    const int wn   = wid >> 2;      // 2 N-strips of 64 cols
    const int l15  = lane & 15;
    const int kbA  = (lane >> 4) * 8;
    const int kbB  = (lane >> 4) * 8;

    const unsigned long long gA = (unsigned long long)(uintptr_t)(A + (size_t)(bm * 128) * K);
    const unsigned long long gB = (unsigned long long)(uintptr_t)(W + (size_t)(bn * 128) * K);

    v8f acc[2][4] = {};

    for (int k0 = 0; k0 < K; k0 += 32) {
        __syncthreads();                         // previous tile fully consumed
#if HAVE_TDM
        if (wid == 0) {
            v4u g0a = tdm_g0(0u,    gA + (unsigned long long)k0 * 2u);
            v4u g0b = tdm_g0(8192u, gB + (unsigned long long)k0 * 2u);
            v8i g1  = tdm_g1(32u, 128u, 32u, 128u, (unsigned long long)K);
            v4i z4  = {};
#if defined(__clang_major__) && __clang_major__ >= 23
            v8i z8 = {};
            __builtin_amdgcn_tensor_load_to_lds(g0a, g1, z4, z4, z8, 0);
            __builtin_amdgcn_tensor_load_to_lds(g0b, g1, z4, z4, z8, 0);
#else
            __builtin_amdgcn_tensor_load_to_lds(g0a, g1, z4, z4, 0);
            __builtin_amdgcn_tensor_load_to_lds(g0b, g1, z4, z4, 0);
#endif
            __builtin_amdgcn_s_wait_tensorcnt((short)0);
        }
#else
        {   // cooperative fallback: 2 x (128 rows x 64 B) in 16-B chunks
            const half_t* gAt = A + (size_t)(bm * 128) * K + k0;
            const half_t* gBt = W + (size_t)(bn * 128) * K + k0;
            for (int c = threadIdx.x; c < 1024; c += 256) {
                const int mat = c >> 9, rc = c & 511;
                const int row = rc >> 2, seg = rc & 3;
                const half_t* src = (mat ? gBt : gAt) + (size_t)row * K + seg * 8;
                _Float16* dst = (mat ? sB : sA) + row * 32 + seg * 8;
                *(uint4*)dst = *(const uint4*)src;
            }
        }
#endif
        __syncthreads();                         // tile visible to all waves

        v16h af[2], bf[4];
#pragma unroll
        for (int t = 0; t < 2; ++t) {
            const _Float16* p = sA + (wm * 32 + t * 16 + l15) * 32 + kbA;
            const v8h lo = *(const v8h*)p;
            const v8h hi = *(const v8h*)(p + 16);
#pragma unroll
            for (int e = 0; e < 8; ++e) { af[t][e] = lo[e]; af[t][e + 8] = hi[e]; }
        }
#pragma unroll
        for (int u = 0; u < 4; ++u) {
            const _Float16* p = sB + (wn * 64 + u * 16 + l15) * 32 + kbB;
            const v8h lo = *(const v8h*)p;
            const v8h hi = *(const v8h*)(p + 16);
#pragma unroll
            for (int e = 0; e < 8; ++e) { bf[u][e] = lo[e]; bf[u][e + 8] = hi[e]; }
        }
#pragma unroll
        for (int t = 0; t < 2; ++t)
#pragma unroll
            for (int u = 0; u < 4; ++u)
                acc[t][u] = __builtin_amdgcn_wmma_f32_16x16x32_f16(
                    false, af[t], false, bf[u], (short)0, acc[t][u], false, false);
    }

    const int nbase = bn * 128 + wn * 64;
#pragma unroll
    for (int u = 0; u < 4; ++u) {
        const int n = nbase + u * 16 + l15;
        const float bv = bias ? bias[n] * biasScale : 0.0f;
#pragma unroll
        for (int t = 0; t < 2; ++t) {
            const int m0 = bm * 128 + wm * 32 + t * 16 + ((lane >> 4) ? 8 : 0);
#pragma unroll
            for (int r = 0; r < 8; ++r) {
                float v = acc[t][u][r] + bv;
                if (doRelu) v = fmaxf(v, 0.0f);
                C[(size_t)(m0 + r) * N + n] = v;
            }
        }
    }
}

// ================= token embedding -> concat[sig|date|asset] ==========
__global__ void embed_kernel(const float* __restrict__ xs, const float* __restrict__ df,
                             const float* __restrict__ ae,
                             const float* __restrict__ sw, const float* __restrict__ sb,
                             const float* __restrict__ dw, const float* __restrict__ db,
                             float* __restrict__ cat)
{
    int idx = blockIdx.x * blockDim.x + threadIdx.x;
    if (idx >= NTOK * DMODEL) return;
    const int c = idx & 511, n = idx >> 9;
    const int d = n & 63, bh = n >> 6;
    const float s0 = xs[n * 2], s1 = xs[n * 2 + 1];
    float* row = cat + (size_t)n * QKVDIM;
    row[c] = sw[c * 2] * s0 + sw[c * 2 + 1] * s1 + sb[c];
    const float* f = df + bh * 3;
    row[512 + c]  = dw[c * 3] * f[0] + dw[c * 3 + 1] * f[1] + dw[c * 3 + 2] * f[2] + db[c];
    row[1024 + c] = ae[d * DMODEL + c];
}

// ============ transpose (B,D,H,C)-view -> (B,H,D,C) contiguous ========
__global__ void transpose_kernel(const float* __restrict__ in, float* __restrict__ out)
{
    int idx = blockIdx.x * blockDim.x + threadIdx.x;
    if (idx >= NTOK * DMODEL) return;
    const int c = idx & 511, n = idx >> 9;
    const int d = n & 63, h = (n >> 6) & 31, b = n >> 11;
    out[idx] = in[(size_t)((b * DD + d) * HH + h) * DMODEL + c];
}

// ======================= attention (per query row) ====================
template <int L, bool CAUSAL>
__global__ void attn_kernel(const float* __restrict__ qkv, float* __restrict__ out,
                            int nseq, const float* __restrict__ maskbuf)
{
    int idx = blockIdx.x * blockDim.x + threadIdx.x;
    if (idx >= nseq * NHEADS * L) return;
    const int i    = idx % L;
    const int head = (idx / L) % NHEADS;
    const int seq  = idx / (L * NHEADS);
    const float* base = qkv + (size_t)seq * L * QKVDIM;
    const float* q = base + (size_t)i * QKVDIM + head * HEADDIM;

    float s[L];
    float mx = -INFINITY;
#pragma unroll
    for (int j = 0; j < L; ++j) {
        if (CAUSAL && j > i) { s[j] = -INFINITY; continue; }
        const float* kr = base + (size_t)j * QKVDIM + DMODEL + head * HEADDIM;
        float dot = 0.f;
        for (int c = 0; c < HEADDIM; ++c) dot += q[c] * kr[c];
        dot *= 0.125f;
        if (!CAUSAL) dot += maskbuf[((seq & 31) * L + i) * L + j];
        s[j] = dot;
        mx = fmaxf(mx, dot);
    }
    float sum = 0.f;
#pragma unroll
    for (int j = 0; j < L; ++j) { const float e = __expf(s[j] - mx); s[j] = e; sum += e; }
    const float inv = 1.f / sum;

    const float* vb = base + 2 * DMODEL + head * HEADDIM;
    float* orow = out + (size_t)(seq * L + i) * DMODEL + head * HEADDIM;
    for (int c = 0; c < HEADDIM; ++c) {
        float a = 0.f;
#pragma unroll
        for (int j = 0; j < L; ++j) a += s[j] * vb[(size_t)j * QKVDIM + c];
        orow[c] = a * inv;
    }
}

// =================== residual + layernorm (wave per row) ==============
__global__ void resln_kernel(const float* x, const float* res,
                             const float* __restrict__ g, const float* __restrict__ b,
                             float* out, int Mrows)
{
    const int row = blockIdx.x * (blockDim.x >> 5) + (threadIdx.x >> 5);
    const int lane = threadIdx.x & 31;
    if (row >= Mrows) return;
    const float* xr = x + (size_t)row * DMODEL;
    const float* rr = res + (size_t)row * DMODEL;
    float v[16];
    float sum = 0.f;
#pragma unroll
    for (int t = 0; t < 16; ++t) { const int c = lane + t * 32; v[t] = xr[c] + rr[c]; sum += v[t]; }
#pragma unroll
    for (int o = 16; o >= 1; o >>= 1) sum += __shfl_xor(sum, o, 32);
    const float mean = sum * (1.0f / 512.0f);
    float var = 0.f;
#pragma unroll
    for (int t = 0; t < 16; ++t) { const float dv = v[t] - mean; var += dv * dv; }
#pragma unroll
    for (int o = 16; o >= 1; o >>= 1) var += __shfl_xor(var, o, 32);
    const float inv = rsqrtf(var * (1.0f / 512.0f) + 1e-5f);
    float* orow = out + (size_t)row * DMODEL;
#pragma unroll
    for (int t = 0; t < 16; ++t) { const int c = lane + t * 32; orow[c] = (v[t] - mean) * inv * g[c] + b[c]; }
}

// === hidsum[bh,j,c] = sum_d relu(w1[c]*cross[bh,d,j] + b1[c]) =========
__global__ void hidsum_kernel(const float* __restrict__ cs,
                              const float* __restrict__ w1, const float* __restrict__ b1,
                              float* __restrict__ out)
{
    int idx = blockIdx.x * blockDim.x + threadIdx.x;
    if (idx >= BB * HH * DD * HIDC) return;
    const int c = idx & 127, j = (idx >> 7) & 63, bh = idx >> 13;
    const float w = w1[c], b = b1[c];
    const float* base = cs + (size_t)bh * DD * DD + j;
    float acc = 0.f;
    for (int d = 0; d < DD; ++d) acc += fmaxf(w * base[d * DD] + b, 0.f);
    out[(size_t)(bh * DD + j) * HIDC + c] = acc;
}

// mask[h,i,j] = softplus(gamma)/32 * sum_b <qe[b,h,i,:], betasum[b,h,j,:]>
__global__ void mask_kernel(const float* __restrict__ qe, const float* __restrict__ bs,
                            const float* __restrict__ gamma, float* __restrict__ mask)
{
    int idx = blockIdx.x * blockDim.x + threadIdx.x;
    if (idx >= HH * DD * DD) return;
    const int j = idx & 63, i = (idx >> 6) & 63, h = idx >> 12;
    float acc = 0.f;
    for (int b = 0; b < BB; ++b) {
        const float* qr = qe + (size_t)((b * HH + h) * DD + i) * HIDC;
        const float* br = bs + (size_t)((b * HH + h) * DD + j) * HIDC;
        for (int c = 0; c < HIDC; ++c) acc += qr[c] * br[c];
    }
    const float g = gamma[0];
    const float sp = (g > 20.f) ? g : log1pf(__expf(g));
    mask[idx] = acc * sp * (1.0f / 32.0f);
}

// =================== ret_proj + softmax over assets ===================
__global__ void head_kernel(const float* __restrict__ tok,
                            const float* __restrict__ rw, const float* __restrict__ rb,
                            float* __restrict__ outp)
{
    __shared__ float mu[DD];
    const int bh = blockIdx.x, t = threadIdx.x;
    const float* row = tok + (size_t)(bh * DD + t) * DMODEL;
    float m = rb[0];
    for (int c = 0; c < DMODEL; ++c) m += row[c] * rw[c];
    mu[t] = m;
    __syncthreads();
    float mx = -INFINITY;
    for (int d = 0; d < DD; ++d) mx = fmaxf(mx, mu[d]);
    float sum = 0.f;
    for (int d = 0; d < DD; ++d) sum += __expf(mu[d] - mx);
    outp[bh * DD + t] = __expf(m - mx) / sum;
    outp[NTOK + bh * DD + t] = m;
}

// =====================================================================
extern "C" void kernel_launch(void* const* d_in, const int* in_sizes, int n_in,
                              void* d_out, int out_size, void* d_ws, size_t ws_size,
                              hipStream_t stream)
{
    (void)in_sizes; (void)n_in; (void)out_size; (void)ws_size;
    auto P = [&](int i) { return (const float*)d_in[i]; };

    const float* x_sigs  = P(0);
    const float* cross   = P(1);
    const float* datef   = P(2);
    const float* a_embed = P(3);
    const float* dp_b = P(4);  const float* dp_w = P(5);
    const float* ip_b = P(6);  const float* ip_w = P(7);
    const int ABASE0 = 8, TBASE0 = 29, LSTRIDE = 33;
    const float* rp_b = P(74); const float* rp_w = P(75);
    const float* se_b = P(76); const float* se_w = P(77);

    char* ws = (char*)d_ws;
    float* P0   = (float*)(ws);
    float* P1   = (float*)(ws + ((size_t)16 << 20));
    float* ATT  = (float*)(ws + ((size_t)32 << 20));
    float* RES  = (float*)(ws + ((size_t)48 << 20));
    float* TMP  = (float*)(ws + ((size_t)64 << 20));
    float* HS   = (float*)(ws + ((size_t)128 << 20));
    float* BS   = (float*)(ws + ((size_t)132 << 20));
    float* QH   = (float*)(ws + ((size_t)136 << 20));
    float* QE   = (float*)(ws + ((size_t)140 << 20));
    float* MSK  = (float*)(ws + ((size_t)144 << 20));
    half_t* HA16 = (half_t*)(ws + ((size_t)148 << 20));   // 32 MB max (8192x2048)
    half_t* HW16 = (half_t*)(ws + ((size_t)180 << 20));   // 4 MB max

    auto gemm = [&](const float* A, const float* Wt, const float* bias, float* C,
                    int M, int N, int K, int relu, float bscale) {
        cvt_f16_kernel<<<(M * K / 2 + 255) / 256, 256, 0, stream>>>(A, HA16, M * K);
        cvt_f16_kernel<<<(N * K / 2 + 255) / 256, 256, 0, stream>>>(Wt, HW16, N * K);
        gemm_wmma_kernel<<<(M / 128) * (N / 128), 256, 16384, stream>>>(
            HA16, HW16, bias, C, M, N, K, relu, bscale);
    };
    const int EW = (NTOK * DMODEL + 255) / 256;

    embed_kernel<<<EW * 3, 256, 0, stream>>>(x_sigs, datef, a_embed, se_w, se_b, dp_w, dp_b, TMP);
    gemm(TMP, ip_w, ip_b, P0, NTOK, DMODEL, QKVDIM, 0, 1.0f);

    float* cur = P0;
    float* alt = P1;
    for (int l = 0; l < 2; ++l) {
        const int A0 = ABASE0 + l * LSTRIDE, T0 = TBASE0 + l * LSTRIDE;
        const float *tff1b=P(T0+0), *tff1w=P(T0+1), *tff2b=P(T0+2), *tff2w=P(T0+3);
        const float *tinb=P(T0+4), *tinw=P(T0+5);
        const float *tl1b=P(T0+6), *tl1g=P(T0+7), *tl2b=P(T0+8), *tl2g=P(T0+9);
        const float *toutb=P(T0+10), *toutw=P(T0+11);
        const float *ab1b=P(A0+0), *ab1w=P(A0+1), *ab2b=P(A0+2), *ab2w=P(A0+3);
        const float *aff1b=P(A0+4), *aff1w=P(A0+5), *aff2b=P(A0+6), *aff2w=P(A0+7);
        const float *agam=P(A0+8);
        const float *ainb=P(A0+9), *ainw=P(A0+10);
        const float *al1b=P(A0+11), *al1g=P(A0+12), *al2b=P(A0+13), *al2g=P(A0+14);
        const float *aoutb=P(A0+15), *aoutw=P(A0+16);
        const float *aq1b=P(A0+17), *aq1w=P(A0+18), *aq2b=P(A0+19), *aq2w=P(A0+20);

        // ---- temporal block (256 seqs x L=32, causal) ----
        gemm(cur, tinw, tinb, TMP, NTOK, QKVDIM, DMODEL, 0, 1.0f);
        attn_kernel<32, true><<<(NTOK * NHEADS) / 256, 256, 0, stream>>>(TMP, ATT, 256, nullptr);
        gemm(ATT, toutw, toutb, RES, NTOK, DMODEL, DMODEL, 0, 1.0f);
        resln_kernel<<<NTOK / 8, 256, 0, stream>>>(RES, cur, tl1g, tl1b, cur, NTOK);
        gemm(cur, tff1w, tff1b, TMP, NTOK, FFDIM, DMODEL, 1, 1.0f);
        gemm(TMP, tff2w, tff2b, RES, NTOK, DMODEL, FFDIM, 0, 1.0f);
        resln_kernel<<<NTOK / 8, 256, 0, stream>>>(RES, cur, tl2g, tl2b, cur, NTOK);

        // ---- reinterpret (B,D,H,C) -> contiguous (B,H,D,C) ----
        transpose_kernel<<<EW, 256, 0, stream>>>(cur, alt);

        // ---- dynamic bias mask[h,i,j] ----
        hidsum_kernel<<<(BB * HH * DD * HIDC) / 256, 256, 0, stream>>>(cross, ab1w, ab1b, HS);
        gemm(HS, ab2w, ab2b, BS, NTOK, HIDC, HIDC, 0, (float)DD);
        gemm(alt, aq1w, aq1b, QH, NTOK, HIDC, DMODEL, 1, 1.0f);
        gemm(QH, aq2w, aq2b, QE, NTOK, HIDC, HIDC, 0, 1.0f);
        mask_kernel<<<(HH * DD * DD) / 256, 256, 0, stream>>>(QE, BS, agam, MSK);

        // ---- asset block (128 seqs x L=64, additive mask) ----
        gemm(alt, ainw, ainb, TMP, NTOK, QKVDIM, DMODEL, 0, 1.0f);
        attn_kernel<64, false><<<(NTOK * NHEADS) / 256, 256, 0, stream>>>(TMP, ATT, 128, MSK);
        gemm(ATT, aoutw, aoutb, RES, NTOK, DMODEL, DMODEL, 0, 1.0f);
        resln_kernel<<<NTOK / 8, 256, 0, stream>>>(RES, alt, al1g, al1b, alt, NTOK);
        gemm(alt, aff1w, aff1b, TMP, NTOK, FFDIM, DMODEL, 1, 1.0f);
        gemm(TMP, aff2w, aff2b, RES, NTOK, DMODEL, FFDIM, 0, 1.0f);
        resln_kernel<<<NTOK / 8, 256, 0, stream>>>(RES, alt, al2g, al2b, alt, NTOK);

        float* t = cur; cur = alt; alt = t;
    }

    head_kernel<<<BB * HH, DD, 0, stream>>>(cur, rp_w, rp_b, (float*)d_out);
}
